// SimilarityTreeLSTM_86904368267879
// MI455X (gfx1250) — matrix operand order repaired and checked by reading, not compiled, not run on hardware
//
#include <hip/hip_runtime.h>
#include <hip/hip_bf16.h>

// ---------------------------------------------------------------------------
// SimilarityTreeLSTM on gfx1250 (wave32, WMMA).
//
// Restructure: level-order TreeLSTM with hoisted h-GEMMs.
//   xiouf[j][0:2048]  = x[j] @ [W_ioux | W_fx]          (precompute, WMMA)
//   per level (deepest->root):
//     elementwise:  iou = xiouf[j] + b + sum_ch gfh[ch][0:1536]
//                   f_m = sigmoid(gfh[ch_m][1536+q] + xiouf[j][1536+q] + b)
//                   c_j = sig(i)*tanh(u) + sum f_m*c[ch_m];  h_j = sig(o)*tanh(c_j)
//     GEMM (WMMA):  gfh[level rows] = h[level rows] @ [W_iouh | W_fh]
//   head: vec=[lc*rc,|lc-rc|] -> sigmoid -> log_softmax (tiny, VALU)
//
// B matrices pre-packed to WMMA lane layout: per (ntile,ktile,lane) 16
// contiguous bf16 = K rows (lane<16: K 0..15, lane>=16: K 16..31) of column
// N = ntile*16 + lane%16.
// ---------------------------------------------------------------------------

typedef __bf16 bf16_t;
typedef __attribute__((ext_vector_type(16))) __bf16 v16bf;
typedef __attribute__((ext_vector_type(8)))  __bf16 v8bf;
typedef __attribute__((ext_vector_type(8)))  float  v8f;

__device__ __forceinline__ float sig_(float x) { return 1.f / (1.f + __expf(-x)); }
__device__ __forceinline__ float tanh_(float x) { return 1.f - 2.f / (__expf(2.f * x) + 1.f); }

// One 16x16 f32 output tile: D[mb..mb+15, nt*16..+15] = A[mb..,0:KT*32] @ Bpacked.
// A row-major bf16 (lda elems). Bpacked per comment above. 16 k-steps of
// v_wmma_f32_16x16x32_bf16 with v8f accumulator.
__device__ __forceinline__ void wmma_tile(const bf16_t* __restrict__ A, int lda,
                                          const bf16_t* __restrict__ Bp, int KT,
                                          float* __restrict__ D, int ldd,
                                          int mb, int nt, int lane) {
  const int l = lane & 15;
  const int hi = lane >> 4;  // 0: K 0..15, 1: K 16..31
  const bf16_t* arow = A + (size_t)(mb + l) * lda + hi * 16;
  const bf16_t* bp = Bp + ((size_t)(nt * KT) * 32 + lane) * 16;
  v8f acc = {};
#pragma unroll 4
  for (int kt = 0; kt < KT; ++kt) {
    v8bf a0 = *(const v8bf*)(arow + kt * 32);
    v8bf a1 = *(const v8bf*)(arow + kt * 32 + 8);
    v16bf a = __builtin_shufflevector(a0, a1, 0, 1, 2, 3, 4, 5, 6, 7, 8, 9, 10, 11, 12, 13, 14, 15);
    v16bf b = *(const v16bf*)(bp + (size_t)kt * 32 * 16);
    acc = __builtin_amdgcn_wmma_f32_16x16x32_bf16(false, a, false, b, (short)0, acc, false, false);
  }
  float* drow = D + (size_t)(mb + hi * 8) * ldd + nt * 16 + l;
#pragma unroll
  for (int r = 0; r < 8; ++r) drow[(size_t)r * ldd] = acc[r];
}

// ---- pack [Wiou(512x1536) | Wf(512x512)] -> bf16 WMMA-B layout (512x2048) ----
__global__ void pack_b_kernel(const float* __restrict__ Wiou, const float* __restrict__ Wf,
                              bf16_t* __restrict__ Bp) {
  int idx = blockIdx.x * blockDim.x + threadIdx.x;  // ((nt*16)+kt)*32+lane
  if (idx >= 128 * 16 * 32) return;
  int lane = idx & 31;
  int n = ((idx >> 9) << 4) + (lane & 15);          // nt*16 + lane%16
  int kbase = (((idx >> 5) & 15) << 5) + ((lane >> 4) << 4);  // kt*32 + (lane>=16)*16
  bf16_t* dst = Bp + (size_t)idx * 16;
#pragma unroll
  for (int e = 0; e < 16; ++e) {
    int k = kbase + e;
    float v = (n < 1536) ? Wiou[(size_t)k * 1536 + n] : Wf[(size_t)k * 512 + (n - 1536)];
    dst[e] = (bf16_t)v;
  }
}

// ---- embedding gather to bf16 + zero the permanent zero rows / pads ----
__global__ void init_kernel(const int* __restrict__ ltok, const int* __restrict__ rtok,
                            const float* __restrict__ emb,
                            bf16_t* __restrict__ x0, bf16_t* __restrict__ x1,
                            bf16_t* __restrict__ h0, bf16_t* __restrict__ h1,
                            float* __restrict__ g0, float* __restrict__ g1,
                            float* __restrict__ c0, float* __restrict__ c1) {
  int idx = blockIdx.x * blockDim.x + threadIdx.x;
  if (idx < 2 * 512 * 512) {  // x gather
    int t = idx >> 18, r = idx & 262143;
    const int* tok = t ? rtok : ltok;
    (t ? x1 : x0)[r] = (bf16_t)emb[(size_t)tok[r >> 9] * 512 + (r & 511)];
    return;
  }
  int j = idx - 2 * 512 * 512;
  if (j < 2 * 16 * 512) {  // h rows 512..527 = 0
    int t = j >= 16 * 512, o = j - t * 16 * 512;
    (t ? h1 : h0)[(size_t)512 * 512 + o] = (bf16_t)0.f;
    return;
  }
  j -= 2 * 16 * 512;
  if (j < 2 * 16 * 2048) {  // gfh rows 512..527 = 0
    int t = j >= 16 * 2048, o = j - t * 16 * 2048;
    (t ? g1 : g0)[(size_t)512 * 2048 + o] = 0.f;
    return;
  }
  j -= 2 * 16 * 2048;
  if (j < 2 * 512) {  // c row 512 = 0
    int t = j >= 512, o = j - t * 512;
    (t ? c1 : c0)[(size_t)512 * 512 + o] = 0.f;
  }
}

// ---- precompute xiouf = x @ [W_ioux | W_fx], both trees, whole chip ----
__global__ void precompute_gemm(const bf16_t* __restrict__ x0, const bf16_t* __restrict__ x1,
                                const bf16_t* __restrict__ Bx,
                                float* __restrict__ d0, float* __restrict__ d1) {
  int w = blockIdx.x * (blockDim.x >> 5) + (threadIdx.x >> 5);
  if (w >= 8192) return;  // uniform per wave -> EXEC stays full for WMMA
  int lane = threadIdx.x & 31;
  int t = w >> 12, tile = w & 4095;
  wmma_tile(t ? x1 : x0, 512, Bx, 16, t ? d1 : d0, 2048, (tile >> 7) << 4, tile & 127, lane);
}

// ---- level-order scan: one workgroup (32 waves) per tree ----
__global__ __launch_bounds__(1024) void scan_kernel(
    const int* __restrict__ lch, const int* __restrict__ rch,
    const float* __restrict__ b_ioux, const float* __restrict__ b_iouh,
    const float* __restrict__ b_fx, const float* __restrict__ b_fh,
    const bf16_t* __restrict__ Bh,
    float* __restrict__ xiouf0, float* __restrict__ xiouf1,
    bf16_t* __restrict__ h0, bf16_t* __restrict__ h1,
    float* __restrict__ gfh0, float* __restrict__ gfh1,
    float* __restrict__ c0, float* __restrict__ c1) {
  const int t = blockIdx.x;
  const int* children = t ? rch : lch;
  float* xiouf = t ? xiouf1 : xiouf0;
  bf16_t* h = t ? h1 : h0;
  float* gfh = t ? gfh1 : gfh0;
  float* c = t ? c1 : c0;
  const int tid = threadIdx.x, lane = tid & 31, wave = tid >> 5;
  const int LB[7] = {0, 171, 427, 491, 507, 511, 512};  // deepest -> root (j-space)

  for (int lvl = 0; lvl < 6; ++lvl) {
    const int lo = LB[lvl], M = LB[lvl + 1] - lo;
    // --- elementwise recurrence for this level's nodes ---
    for (int e = tid; e < M * 512; e += 1024) {
      const int j = lo + (e >> 9), q = e & 511;
      const int k0 = children[j * 4 + 0], k1 = children[j * 4 + 1];
      const int k2 = children[j * 4 + 2], k3 = children[j * 4 + 3];
      const float* g0 = gfh + (size_t)k0 * 2048;
      const float* g1 = gfh + (size_t)k1 * 2048;
      const float* g2 = gfh + (size_t)k2 * 2048;
      const float* g3 = gfh + (size_t)k3 * 2048;
      const float* xr = xiouf + (size_t)j * 2048;
      float ig = sig_(xr[q] + g0[q] + g1[q] + g2[q] + g3[q] + b_ioux[q] + b_iouh[q]);
      float og = sig_(xr[512 + q] + g0[512 + q] + g1[512 + q] + g2[512 + q] + g3[512 + q] +
                      b_ioux[512 + q] + b_iouh[512 + q]);
      float ug = tanh_(xr[1024 + q] + g0[1024 + q] + g1[1024 + q] + g2[1024 + q] + g3[1024 + q] +
                       b_ioux[1024 + q] + b_iouh[1024 + q]);
      float fb = xr[1536 + q] + b_fx[q] + b_fh[q];
      float fc = sig_(g0[1536 + q] + fb) * c[(size_t)k0 * 512 + q] +
                 sig_(g1[1536 + q] + fb) * c[(size_t)k1 * 512 + q] +
                 sig_(g2[1536 + q] + fb) * c[(size_t)k2 * 512 + q] +
                 sig_(g3[1536 + q] + fb) * c[(size_t)k3 * 512 + q];
      float cj = ig * ug + fc;
      c[(size_t)j * 512 + q] = cj;
      h[(size_t)j * 512 + q] = (bf16_t)(og * tanh_(cj));
    }
    __syncthreads();
    if (lvl < 5) {  // root's gfh is never consumed
      // --- gfh[level rows] = h[level rows] @ [W_iouh | W_fh] via WMMA ---
      const int tiles = ((M + 15) >> 4) * 128;
      for (int tl = wave; tl < tiles; tl += 32)
        wmma_tile(h, 512, Bh, 16, gfh, 2048, lo + ((tl >> 7) << 4), tl & 127, lane);
      __syncthreads();
    }
  }
}

// ---- similarity head: vec=[lc*rc,|lc-rc|] -> sigmoid MLP -> log_softmax ----
__global__ __launch_bounds__(512) void head_kernel(
    const float* __restrict__ c0, const float* __restrict__ c1,
    const float* __restrict__ Wh, const float* __restrict__ bh,
    const float* __restrict__ Wp, const float* __restrict__ bp,
    float* __restrict__ out) {
  __shared__ float hid[512];
  __shared__ float logit[5];
  const int tid = threadIdx.x;
  const float* lc = c0 + (size_t)511 * 512;
  const float* rc = c1 + (size_t)511 * 512;
  float acc = bh[tid];
  for (int q = 0; q < 512; ++q) {
    float a = lc[q], b = rc[q];
    acc += (a * b) * Wh[(size_t)q * 512 + tid] + fabsf(a - b) * Wh[(size_t)(512 + q) * 512 + tid];
  }
  hid[tid] = sig_(acc);
  __syncthreads();
  if (tid < 5) {
    float s = bp[tid];
    for (int k = 0; k < 512; ++k) s += hid[k] * Wp[k * 5 + tid];
    logit[tid] = s;
  }
  __syncthreads();
  if (tid == 0) {
    float mx = logit[0];
    for (int i = 1; i < 5; ++i) mx = fmaxf(mx, logit[i]);
    float se = 0.f;
    for (int i = 0; i < 5; ++i) se += __expf(logit[i] - mx);
    float lse = __logf(se) + mx;
    for (int i = 0; i < 5; ++i) out[i] = logit[i] - lse;
  }
}

extern "C" void kernel_launch(void* const* d_in, const int* in_sizes, int n_in,
                              void* d_out, int out_size, void* d_ws, size_t ws_size,
                              hipStream_t stream) {
  const int* linputs = (const int*)d_in[0];
  const int* rinputs = (const int*)d_in[1];
  const int* lch = (const int*)d_in[2];
  const int* rch = (const int*)d_in[3];
  const float* emb = (const float*)d_in[4];
  const float* W_ioux = (const float*)d_in[5];
  const float* b_ioux = (const float*)d_in[6];
  const float* W_iouh = (const float*)d_in[7];
  const float* b_iouh = (const float*)d_in[8];
  const float* W_fx = (const float*)d_in[9];
  const float* b_fx = (const float*)d_in[10];
  const float* W_fh = (const float*)d_in[11];
  const float* b_fh = (const float*)d_in[12];
  const float* Wh = (const float*)d_in[13];
  const float* bh = (const float*)d_in[14];
  const float* Wp = (const float*)d_in[15];
  const float* bp = (const float*)d_in[16];

  char* ws = (char*)d_ws;
  size_t off = 0;
  auto alloc = [&](size_t bytes) {
    void* p = ws + off;
    off = (off + bytes + 255) & ~(size_t)255;
    return p;
  };
  bf16_t* Bx = (bf16_t*)alloc((size_t)512 * 2048 * 2);   // [W_ioux|W_fx] packed
  bf16_t* Bh = (bf16_t*)alloc((size_t)512 * 2048 * 2);   // [W_iouh|W_fh] packed
  bf16_t* x0 = (bf16_t*)alloc((size_t)512 * 512 * 2);
  bf16_t* x1 = (bf16_t*)alloc((size_t)512 * 512 * 2);
  float* xiouf0 = (float*)alloc((size_t)512 * 2048 * 4);
  float* xiouf1 = (float*)alloc((size_t)512 * 2048 * 4);
  bf16_t* h0 = (bf16_t*)alloc((size_t)528 * 512 * 2);    // 513 rows + tile pad
  bf16_t* h1 = (bf16_t*)alloc((size_t)528 * 512 * 2);
  float* gfh0 = (float*)alloc((size_t)528 * 2048 * 4);
  float* gfh1 = (float*)alloc((size_t)528 * 2048 * 4);
  float* c0 = (float*)alloc((size_t)513 * 512 * 4);
  float* c1 = (float*)alloc((size_t)513 * 512 * 4);
  (void)in_sizes; (void)n_in; (void)out_size; (void)ws_size;

  // 1) gather embeddings -> bf16, zero permanent-zero rows and pads
  {
    int total = 2 * 512 * 512 + 2 * 16 * 512 + 2 * 16 * 2048 + 2 * 512;
    init_kernel<<<(total + 255) / 256, 256, 0, stream>>>(linputs, rinputs, emb, x0, x1, h0, h1,
                                                         gfh0, gfh1, c0, c1);
  }
  // 2) pack weight matrices into WMMA-B lane layout (bf16)
  pack_b_kernel<<<256, 256, 0, stream>>>(W_ioux, W_fx, Bx);
  pack_b_kernel<<<256, 256, 0, stream>>>(W_iouh, W_fh, Bh);
  // 3) precompute xiouf = x @ [W_ioux|W_fx] for both trees (8192 wave-tiles)
  precompute_gemm<<<1024, 256, 0, stream>>>(x0, x1, Bx, xiouf0, xiouf1);
  // 4) level-order TreeLSTM scan, one workgroup per tree
  scan_kernel<<<2, 1024, 0, stream>>>(lch, rch, b_ioux, b_iouh, b_fx, b_fh, Bh, xiouf0, xiouf1,
                                      h0, h1, gfh0, gfh1, c0, c1);
  // 5) similarity head -> log_softmax [1,5]
  head_kernel<<<1, 512, 0, stream>>>(c0, c1, Wh, bh, Wp, bp, (float*)d_out);
}
